// BiMambaEncoder_56710748176551
// MI455X (gfx1250) — matrix operand order repaired and checked by reading, not compiled, not run on hardware
//
#include <hip/hip_runtime.h>
#include <stdint.h>

#define NL 2
#define DMODEL 512
#define DINNER 1024
#define DCONV 4
#define DTRANK 32
#define DSTATE 16
#define DFF 1024
#define BATCHSZ 4
#define SEQLEN 2048
#define ROWS (BATCHSZ*SEQLEN)   // 8192

typedef __attribute__((ext_vector_type(16))) __bf16 v16bf;
typedef __attribute__((ext_vector_type(8)))  __bf16 v8bf;
typedef __attribute__((ext_vector_type(8)))  float  v8f;

// ---------------- device helpers ----------------

__device__ __forceinline__ uint16_t f32_to_bf16(float f){
  uint32_t u = __builtin_bit_cast(uint32_t, f);
  if ((u & 0x7F800000u) == 0x7F800000u) return (uint16_t)(u >> 16); // inf/nan passthrough
  uint32_t r = u + 0x7FFFu + ((u >> 16) & 1u);                      // round-to-nearest-even
  return (uint16_t)(r >> 16);
}

__device__ __forceinline__ float siluf(float x){ return x / (1.f + __expf(-x)); }
__device__ __forceinline__ float softplusf(float x){ return (x > 20.f) ? x : log1pf(__expf(x)); }

// A fragment: 16x32 bf16, lane L<16 holds row M=L with K in {0..7,16..23};
// lanes 16..31 hold K in {8..15,24..31}. p points at &A[m0+row][k + kb].
__device__ __forceinline__ v16bf load_fragA(const uint16_t* p){
  v8bf lo = *(const v8bf*)(p);
  v8bf hi = *(const v8bf*)(p + 16);
  v16bf r;
#pragma unroll
  for (int i = 0; i < 8; ++i){ r[i] = lo[i]; r[i+8] = hi[i]; }
  return r;
}

// gfx1250 async copy global -> LDS, 16B per lane, tracked by ASYNCcnt
__device__ __forceinline__ void async_load_b128(uint32_t lds_off, const void* gptr){
  asm volatile("global_load_async_to_lds_b128 %0, %1, off"
               :: "v"(lds_off), "v"((uint64_t)(uintptr_t)gptr)
               : "memory");
}

// ---------------- GEMM: C[M,N] = act(A[M,K] * W[N,K]^T + bias) ----------------
// A, W bf16 row-major; C f32. Block: 256 threads = 8 waves; block tile 128(M) x 64(N);
// wave tile 16(M) x 64(N) = 4 WMMA accumulators.
// B (weight) 64x32 k-tiles are staged into LDS once per block via
// global_load_async_to_lds_b128, double-buffered (ASYNCcnt + barriers),
// removing the 8x redundant per-wave B traffic.
__global__ void __launch_bounds__(256)
gemm_wmma_bf16(const uint16_t* __restrict__ A, int lda,
               const uint16_t* __restrict__ W, int ldw,
               const float* __restrict__ bias,
               float* __restrict__ C, int ldc,
               int K, int act)
{
  __shared__ uint16_t ldsB[2][64*32];   // 2 x 4 KB double buffer

  const int tid  = threadIdx.x;
  const int lane = tid & 31;
  const int wave = tid >> 5;
  const int m0 = blockIdx.y * 128 + wave * 16;
  const int n0 = blockIdx.x * 64;
  const int r15 = lane & 15;
  const int hi  = lane >> 4;
  const int kbA = hi * 8;    // A per-lane K base

  // cooperative staging assignment: thread -> (row 0..63, 16B chunk 0..3)
  const int srow = tid >> 2;
  const int schk = tid & 3;
  const uint16_t* Wst = W + (size_t)(n0 + srow) * ldw + schk * 8;
  const uint32_t ldsOff0 = (uint32_t)(uintptr_t)&ldsB[0][srow*32 + schk*8];
  const uint32_t ldsOff1 = (uint32_t)(uintptr_t)&ldsB[1][srow*32 + schk*8];

  const uint16_t* Arow = A + (size_t)(m0 + r15) * lda + kbA;

  v8f acc[4] = {};
  const int iters = K >> 5;

  async_load_b128(ldsOff0, Wst);                 // k=0 tile -> buf0

  for (int it = 0; it < iters; ++it){
    const int k = it << 5;
    const bool hasNext = (it + 1 < iters);
    if (hasNext)
      async_load_b128((it & 1) ? ldsOff0 : ldsOff1, Wst + k + 32);  // prefetch next tile
    if (hasNext) asm volatile("s_wait_asynccnt 0x1" ::: "memory");  // current tile landed
    else         asm volatile("s_wait_asynccnt 0x0" ::: "memory");
    __syncthreads();                                                // all waves' stores visible

    __builtin_prefetch(Arow + k + 32, 0, 0);                        // global_prefetch_b8 on A
    v16bf a = load_fragA(Arow + k);
    const uint16_t* Bbuf = ldsB[it & 1];
#pragma unroll
    for (int t = 0; t < 4; ++t){
      // B frag from LDS: lane holds N = t*16 + r15, K in {hi*8..hi*8+7, 16+hi*8..16+hi*8+7} (ds_load_b128 x2)
      const v8bf blo = *(const v8bf*)(Bbuf + (t*16 + r15)*32 + hi*8);
      const v8bf bhi = *(const v8bf*)(Bbuf + (t*16 + r15)*32 + hi*8 + 16);
      v16bf b;
#pragma unroll
      for (int i = 0; i < 8; ++i){ b[i] = blo[i]; b[i+8] = bhi[i]; }
      acc[t] = __builtin_amdgcn_wmma_f32_16x16x32_bf16(false, a, false, b, (short)0, acc[t], false, false);
    }
    __syncthreads();                              // buffer consumed before it is overwritten
  }

#pragma unroll
  for (int t = 0; t < 4; ++t){
    const int n = n0 + t*16 + r15;
    const float bia = bias ? bias[n] : 0.f;
#pragma unroll
    for (int r = 0; r < 8; ++r){
      float v = acc[t][r] + bia;
      if (act == 1) v = fmaxf(v, 0.f);
      else if (act == 2) v = softplusf(v);
      C[(size_t)(m0 + 8*hi + r) * ldc + n] = v;   // C layout: VGPR r -> M = r + 8*(lane>=16)
    }
  }
}

// ---------------- elementwise / conv / scan / LN ----------------

__global__ void __launch_bounds__(256)
cvt_bf16_kernel(const float* __restrict__ in, uint16_t* __restrict__ out, int n){
  int i = blockIdx.x*256 + threadIdx.x;
  if (i < n) out[i] = f32_to_bf16(in[i]);
}

__global__ void __launch_bounds__(256)
add2_kernel(const float* __restrict__ a, const float* __restrict__ b, float* __restrict__ o, int n){
  int i = blockIdx.x*256 + threadIdx.x;
  if (i < n) o[i] = a[i] + b[i];
}

// depthwise causal conv (width 4) + bias + SiLU on xc = xz[:, 0:1024]
__global__ void __launch_bounds__(256)
conv_silu_kernel(const float* __restrict__ xz, const float* __restrict__ cw,
                 const float* __restrict__ cb, float* __restrict__ xc,
                 uint16_t* __restrict__ xcbf)
{
  int idx = blockIdx.x*256 + threadIdx.x;         // over ROWS*DINNER
  int d  = idx & (DINNER-1);
  int rl = idx >> 10;
  int l  = rl & (SEQLEN-1);
  int b  = rl >> 11;
  float s = cb[d];
#pragma unroll
  for (int t = 0; t < DCONV; ++t){
    int ls = l - (DCONV-1) + t;
    if (ls >= 0) s = fmaf(cw[d*DCONV + t], xz[(size_t)(b*SEQLEN + ls)*(2*DINNER) + d], s);
  }
  float o = siluf(s);
  xc[idx] = o;
  xcbf[idx] = f32_to_bf16(o);
}

// selective scan: thread per (batch, channel); h[16] in registers; B/C via LDS broadcast
__global__ void __launch_bounds__(256)
scan_kernel(const float* __restrict__ delta, const float* __restrict__ xc,
            const float* __restrict__ dbc, const float* __restrict__ Alog,
            const float* __restrict__ Dp, float* __restrict__ y)
{
  __shared__ float sBC[32];
  const int b = blockIdx.y;
  const int d = blockIdx.x*256 + threadIdx.x;
  float A[DSTATE];
#pragma unroll
  for (int s = 0; s < DSTATE; ++s) A[s] = -__expf(Alog[d*DSTATE + s]);
  const float dp = Dp[d];
  float h[DSTATE] = {};
  for (int l = 0; l < SEQLEN; ++l){
    const size_t r = (size_t)b*SEQLEN + l;
    __syncthreads();
    if (threadIdx.x < 32) sBC[threadIdx.x] = dbc[r*64 + 32 + threadIdx.x];
    __syncthreads();
    const float dt = delta[r*DINNER + d];
    const float xt = xc[r*DINNER + d];
    const float dx = dt * xt;
    float acc = 0.f;
#pragma unroll
    for (int s = 0; s < DSTATE; ++s){
      float dA = __expf(dt * A[s]);
      h[s] = fmaf(dA, h[s], dx * sBC[s]);
      acc = fmaf(h[s], sBC[16 + s], acc);
    }
    y[r*DINNER + d] = fmaf(dp, xt, acc);
  }
}

// ybf = bf16( y * silu(z) ), z = xz[:, 1024:2048]
__global__ void __launch_bounds__(256)
ymul_kernel(const float* __restrict__ y, const float* __restrict__ xz, uint16_t* __restrict__ ybf)
{
  int idx = blockIdx.x*256 + threadIdx.x;
  int d = idx & (DINNER-1);
  size_t r = (size_t)(idx >> 10);
  float z = xz[r*(2*DINNER) + DINNER + d];
  ybf[idx] = f32_to_bf16(y[idx] * siluf(z));
}

// reverse sequence dimension of a (B,SEQ,DMODEL) tensor
__global__ void __launch_bounds__(256)
flip_kernel(const float* __restrict__ in, float* __restrict__ out)
{
  int idx = blockIdx.x*256 + threadIdx.x;         // over ROWS*DMODEL
  int c  = idx & (DMODEL-1);
  int rl = idx >> 9;
  int l  = rl & (SEQLEN-1);
  int b  = rl >> 11;
  out[idx] = in[((size_t)b*SEQLEN + (SEQLEN-1-l))*DMODEL + c];
}

// out = LN(a+b)*g + beta ; optional bf16 mirror
__global__ void __launch_bounds__(128)
add_ln_kernel(const float* __restrict__ a, const float* __restrict__ b,
              const float* __restrict__ g, const float* __restrict__ be,
              float* __restrict__ out, uint16_t* __restrict__ outbf)
{
  __shared__ float red[128];
  const int row = blockIdx.x;
  const int t = threadIdx.x;
  const size_t base = (size_t)row * DMODEL;
  float v[4];
  float s = 0.f;
#pragma unroll
  for (int j = 0; j < 4; ++j){
    int c = t + j*128;
    v[j] = a[base + c] + b[base + c];
    s += v[j];
  }
  red[t] = s; __syncthreads();
  for (int off = 64; off > 0; off >>= 1){ if (t < off) red[t] += red[t+off]; __syncthreads(); }
  const float mean = red[0] * (1.f/DMODEL);
  __syncthreads();
  float s2 = 0.f;
#pragma unroll
  for (int j = 0; j < 4; ++j){ float dd = v[j]-mean; s2 += dd*dd; }
  red[t] = s2; __syncthreads();
  for (int off = 64; off > 0; off >>= 1){ if (t < off) red[t] += red[t+off]; __syncthreads(); }
  const float inv = rsqrtf(red[0]*(1.f/DMODEL) + 1e-5f);
#pragma unroll
  for (int j = 0; j < 4; ++j){
    int c = t + j*128;
    float o = (v[j]-mean)*inv*g[c] + be[c];
    out[base + c] = o;
    if (outbf) outbf[base + c] = f32_to_bf16(o);
  }
}

// ---------------- host driver ----------------

extern "C" void kernel_launch(void* const* d_in, const int* in_sizes, int n_in,
                              void* d_out, int out_size, void* d_ws, size_t ws_size,
                              hipStream_t stream)
{
  (void)in_sizes; (void)n_in; (void)out_size; (void)ws_size;

  const float* x_in   = (const float*)d_in[0];
  const float* Win    = (const float*)d_in[1];
  const float* conv_w = (const float*)d_in[2];
  const float* conv_b = (const float*)d_in[3];
  const float* Wx     = (const float*)d_in[4];
  const float* Wdt    = (const float*)d_in[5];
  const float* bdt    = (const float*)d_in[6];
  const float* A_log  = (const float*)d_in[7];
  const float* Dp     = (const float*)d_in[8];
  const float* Wout   = (const float*)d_in[9];
  const float* ln_g   = (const float*)d_in[10];
  const float* ln_b   = (const float*)d_in[11];
  const float* W1     = (const float*)d_in[12];
  const float* b1     = (const float*)d_in[13];
  const float* W2     = (const float*)d_in[14];
  const float* b2     = (const float*)d_in[15];

  char* base = (char*)d_ws;
  size_t off = 0;
  auto alloc = [&](size_t bytes){ void* p = base + off; off = (off + bytes + 255) & ~(size_t)255; return p; };

  const size_t nWin  = (size_t)NL*2*(2*DINNER)*DMODEL;
  const size_t nWx   = (size_t)NL*2*(DTRANK+2*DSTATE)*DINNER;
  const size_t nWdt  = (size_t)NL*2*DINNER*DTRANK;
  const size_t nWout = (size_t)NL*2*DMODEL*DINNER;
  const size_t nW1   = (size_t)NL*2*DFF*DMODEL;
  const size_t nW2   = (size_t)NL*2*DMODEL*DFF;

  uint16_t* Winbf  = (uint16_t*)alloc(nWin*2);
  uint16_t* Wxbf   = (uint16_t*)alloc(nWx*2);
  uint16_t* Wdtbf  = (uint16_t*)alloc(nWdt*2);
  uint16_t* Woutbf = (uint16_t*)alloc(nWout*2);
  uint16_t* W1bf   = (uint16_t*)alloc(nW1*2);
  uint16_t* W2bf   = (uint16_t*)alloc(nW2*2);

  float*    xz      = (float*)   alloc((size_t)ROWS*2*DINNER*4);
  float*    xc_act  = (float*)   alloc((size_t)ROWS*DINNER*4);
  uint16_t* xc_bf   = (uint16_t*)alloc((size_t)ROWS*DINNER*2);
  float*    dbc     = (float*)   alloc((size_t)ROWS*64*4);
  uint16_t* dbc_bf  = (uint16_t*)alloc((size_t)ROWS*64*2);
  float*    delta   = (float*)   alloc((size_t)ROWS*DINNER*4);
  float*    yscan   = (float*)   alloc((size_t)ROWS*DINNER*4);
  uint16_t* ybf     = (uint16_t*)alloc((size_t)ROWS*DINNER*2);
  float*    mb_out  = (float*)   alloc((size_t)ROWS*DMODEL*4);
  float*    t1f     = (float*)   alloc((size_t)ROWS*DMODEL*4);
  float*    x_cur   = (float*)   alloc((size_t)ROWS*DMODEL*4);
  uint16_t* x_bf    = (uint16_t*)alloc((size_t)ROWS*DMODEL*2);
  float*    xf      = (float*)   alloc((size_t)ROWS*DMODEL*4);
  uint16_t* xf_bf   = (uint16_t*)alloc((size_t)ROWS*DMODEL*2);
  float*    xf2     = (float*)   alloc((size_t)ROWS*DMODEL*4);
  uint16_t* xf2_bf  = (uint16_t*)alloc((size_t)ROWS*DMODEL*2);
  float*    xb      = (float*)   alloc((size_t)ROWS*DMODEL*4);
  float*    xb2     = (float*)   alloc((size_t)ROWS*DMODEL*4);
  float*    xflip   = (float*)   alloc((size_t)ROWS*DMODEL*4);
  uint16_t* xflip_bf= (uint16_t*)alloc((size_t)ROWS*DMODEL*2);

  // FFN scratch aliases dead mamba scratch (mamba block fully finished before FFN runs)
  float*    ffn_h   = xc_act;     // ROWS x DFF
  uint16_t* h_bf    = xc_bf;      // ROWS x DFF
  float*    ffn_out = yscan;      // ROWS x DMODEL (<= ROWS x DINNER)

  auto cvt = [&](const float* s, uint16_t* dst, size_t n){
    cvt_bf16_kernel<<<dim3((unsigned)((n + 255)/256)), dim3(256), 0, stream>>>(s, dst, (int)n);
  };
  auto gemm = [&](const uint16_t* A, int lda, const uint16_t* W, int ldw, const float* bias,
                  float* C, int ldc, int M, int N, int K, int act){
    gemm_wmma_bf16<<<dim3(N/64, M/128), dim3(256), 0, stream>>>(A, lda, W, ldw, bias, C, ldc, K, act);
  };
  auto addln = [&](const float* a, const float* b, const float* g, const float* be,
                   float* o, uint16_t* obf){
    add_ln_kernel<<<dim3(ROWS), dim3(128), 0, stream>>>(a, b, g, be, o, obf);
  };
  auto flip = [&](const float* in, float* out){
    flip_kernel<<<dim3(ROWS*DMODEL/256), dim3(256), 0, stream>>>(in, out);
  };

  // ---- convert all weights to bf16 once ----
  cvt(Win,  Winbf,  nWin);
  cvt(Wx,   Wxbf,   nWx);
  cvt(Wdt,  Wdtbf,  nWdt);
  cvt(Wout, Woutbf, nWout);
  cvt(W1,   W1bf,   nW1);
  cvt(W2,   W2bf,   nW2);

  // ---- x = input ----
  hipMemcpyAsync(x_cur, x_in, (size_t)ROWS*DMODEL*4, hipMemcpyDeviceToDevice, stream);
  cvt(x_cur, x_bf, (size_t)ROWS*DMODEL);

  auto run_mamba = [&](int i, int dir, const uint16_t* xin_bf, float* outbuf){
    const int ld = i*2 + dir;
    // 1) xz = x @ Win^T  (M=8192, N=2048, K=512)
    gemm(xin_bf, DMODEL, Winbf + (size_t)ld*(2*DINNER)*DMODEL, DMODEL, nullptr,
         xz, 2*DINNER, ROWS, 2*DINNER, DMODEL, 0);
    // 2) depthwise causal conv + SiLU
    conv_silu_kernel<<<dim3(ROWS*DINNER/256), dim3(256), 0, stream>>>(
        xz, conv_w + (size_t)ld*DINNER*DCONV, conv_b + (size_t)ld*DINNER, xc_act, xc_bf);
    // 3) dbc = xc @ Wx^T  (N=64, K=1024)
    gemm(xc_bf, DINNER, Wxbf + (size_t)ld*(DTRANK+2*DSTATE)*DINNER, DINNER, nullptr,
         dbc, 64, ROWS, 64, DINNER, 0);
    cvt(dbc, dbc_bf, (size_t)ROWS*64);
    // 4) delta = softplus(dbc[:, :32] @ Wdt^T + bdt)  (N=1024, K=32, lda=64)
    gemm(dbc_bf, 64, Wdtbf + (size_t)ld*DINNER*DTRANK, DTRANK, bdt + (size_t)ld*DINNER,
         delta, DINNER, ROWS, DINNER, DTRANK, 2);
    // 5) selective scan + Dp*xc
    scan_kernel<<<dim3(DINNER/256, BATCHSZ), dim3(256), 0, stream>>>(
        delta, xc_act, dbc, A_log + (size_t)ld*DINNER*DSTATE, Dp + (size_t)ld*DINNER, yscan);
    // 6) y *= silu(z)  -> bf16
    ymul_kernel<<<dim3(ROWS*DINNER/256), dim3(256), 0, stream>>>(yscan, xz, ybf);
    // 7) out = y @ Wout^T  (N=512, K=1024)
    gemm(ybf, DINNER, Woutbf + (size_t)ld*DMODEL*DINNER, DINNER, nullptr,
         outbuf, DMODEL, ROWS, DMODEL, DINNER, 0);
  };

  for (int i = 0; i < NL; ++i){
    const float* g0 = ln_g + ((size_t)i*4 + 0)*DMODEL; const float* be0 = ln_b + ((size_t)i*4 + 0)*DMODEL;
    const float* g1 = ln_g + ((size_t)i*4 + 1)*DMODEL; const float* be1 = ln_b + ((size_t)i*4 + 1)*DMODEL;
    const float* g2 = ln_g + ((size_t)i*4 + 2)*DMODEL; const float* be2 = ln_b + ((size_t)i*4 + 2)*DMODEL;
    const float* g3 = ln_g + ((size_t)i*4 + 3)*DMODEL; const float* be3 = ln_b + ((size_t)i*4 + 3)*DMODEL;

    // forward branch: x_f = LN(mb0(x) + x)
    run_mamba(i, 0, x_bf, mb_out);
    addln(mb_out, x_cur, g0, be0, xf, xf_bf);

    // x_f = LN(ffn0(x_f) + x_f)
    gemm(xf_bf, DMODEL, W1bf + (size_t)(i*2+0)*DFF*DMODEL, DMODEL, b1 + (size_t)(i*2+0)*DFF,
         ffn_h, DFF, ROWS, DFF, DMODEL, 1 /*relu*/);
    cvt(ffn_h, h_bf, (size_t)ROWS*DFF);
    gemm(h_bf, DFF, W2bf + (size_t)(i*2+0)*DMODEL*DFF, DFF, b2 + (size_t)(i*2+0)*DMODEL,
         ffn_out, DMODEL, ROWS, DMODEL, DFF, 0);
    addln(ffn_out, xf, g1, be1, xf2, xf2_bf);

    // backward branch: x_b = LN(flip(mb1(flip(x))) + x)
    flip(x_cur, xflip);
    cvt(xflip, xflip_bf, (size_t)ROWS*DMODEL);
    run_mamba(i, 1, xflip_bf, mb_out);
    flip(mb_out, t1f);
    addln(t1f, x_cur, g2, be2, xb, nullptr);

    // x_b = LN(ffn1(x_f) + x_b)   (ffn1 applied to post-FFN0 x_f!)
    gemm(xf2_bf, DMODEL, W1bf + (size_t)(i*2+1)*DFF*DMODEL, DMODEL, b1 + (size_t)(i*2+1)*DFF,
         ffn_h, DFF, ROWS, DFF, DMODEL, 1 /*relu*/);
    cvt(ffn_h, h_bf, (size_t)ROWS*DFF);
    gemm(h_bf, DFF, W2bf + (size_t)(i*2+1)*DMODEL*DFF, DFF, b2 + (size_t)(i*2+1)*DMODEL,
         ffn_out, DMODEL, ROWS, DMODEL, DFF, 0);
    addln(ffn_out, xb, g3, be3, xb2, nullptr);

    // x = x_f + x_b
    add2_kernel<<<dim3(ROWS*DMODEL/256), dim3(256), 0, stream>>>(xf2, xb2, x_cur, ROWS*DMODEL);
    cvt(x_cur, x_bf, (size_t)ROWS*DMODEL);
  }

  hipMemcpyAsync(d_out, x_cur, (size_t)ROWS*DMODEL*4, hipMemcpyDeviceToDevice, stream);
}
